// CRF_10625749090976
// MI455X (gfx1250) — compile-verified, matrix operand order checked by reading, not run on hardware
//
#include <hip/hip_runtime.h>

// CRF forward (log-partition) for B=512, T=512, K=128 on gfx1250.
// logsumexp recurrence rewritten as f16 WMMA GEMM with f32 accumulate:
//   alpha' = feat + m + log( exp(alpha - m) @ exp(trans) )
typedef __attribute__((ext_vector_type(16))) _Float16 v16h;
typedef __attribute__((ext_vector_type(8)))  _Float16 v8h;
typedef __attribute__((ext_vector_type(8)))  float    v8f;

#define KTAG 128
#define TSZ  512
#define NEGV -10000.0f
#define START_TAG 126
#define END_TAG   127
#define LOG2E 1.4426950408889634f
#define LN2   0.6931471805599453f

__device__ __forceinline__ float fast_exp(float x) {  // v_exp_f32 based
    return __builtin_amdgcn_exp2f(x * LOG2E);
}
__device__ __forceinline__ float fast_log(float x) {  // v_log_f32 based
    return __builtin_amdgcn_logf(x) * LN2;
}

__global__ __launch_bounds__(256) void crf_forward_kernel(
    const float* __restrict__ feats,     // [B,T,K]
    const float* __restrict__ trans,     // [K,K]
    const int*   __restrict__ lengths,   // [B], sorted descending
    float*       __restrict__ partial)   // [gridDim.x] block partial sums
{
    __shared__ __align__(16) float    s_alpha[16][132];  // padded pitch
    __shared__ __align__(16) _Float16 s_P[16][136];      // exp(alpha-m), padded pitch
    __shared__ __align__(16) float    s_mrow[16];
    __shared__ int   s_len[16];
    __shared__ float s_res[16];

    const int tid  = threadIdx.x;
    const int wave = tid >> 5;          // 0..7  -> output column tile
    const int lane = tid & 31;
    const int half = lane >> 4;         // 0/1: K-half / +8 row offset in C/D
    const int l16  = lane & 15;
    const int b0   = blockIdx.x * 16;
    const int n0   = wave * 16;
    const int j    = n0 + l16;          // this lane's output column (C/D: lane = N)

    const int row16 = tid >> 4;         // helper mapping: 16 threads per row
    const int c16   = tid & 15;
    const int cbase = c16 * 8;          // 8 contiguous columns per thread

    if (tid < 16) s_len[tid] = lengths[b0 + tid];
    for (int e = 0; e < 8; ++e) {
        int c = cbase + e;
        s_alpha[row16][c] = (c == START_TAG) ? 0.0f : NEGV;
    }
    __syncthreads();

    // ---- Preload loop-invariant B fragments: ET = exp(trans), f16 -------
    // B (32x16 per WMMA): lane = N column; element e<8  -> K = kk + half*8 + e
    //                                      element e>=8 -> K = kk + half*8 + 16 + (e - 8)
    v16h Bfrag[4];
    for (int kc = 0; kc < 4; ++kc) {
        int base = kc * 32 + half * 8;
        for (int e = 0; e < 8; ++e) {
            Bfrag[kc][e]     = (_Float16)__expf(trans[(base + e)      * KTAG + j]);
            Bfrag[kc][e + 8] = (_Float16)__expf(trans[(base + 16 + e) * KTAG + j]);
        }
    }

    int lenr[8];
    for (int r = 0; r < 8; ++r) lenr[r] = s_len[r + 8 * half];
    int maxlen = s_len[0];
    for (int r = 1; r < 16; ++r) maxlen = max(maxlen, s_len[r]);  // uniform

    const _Float16* prow = &s_P[l16][0];  // A: lane = M row

    // ---------------- time recurrence ----------------
    for (int t = 0; t < maxlen; ++t) {
        // emissions for this wave's D tile (row m = r + 8*half, col j); issue early
        float fv[8];
        {
            const float* fb = feats + ((size_t)(b0 + 8 * half) * TSZ + (size_t)t) * KTAG + j;
            for (int r = 0; r < 8; ++r) fv[r] = fb[(size_t)r * TSZ * KTAG];
        }

        // per-row max of alpha: 8 contiguous cols + half-wave shuffle reduce
        float4 av0 = *(const float4*)&s_alpha[row16][cbase];
        float4 av1 = *(const float4*)&s_alpha[row16][cbase + 4];
        float pm = fmaxf(fmaxf(fmaxf(av0.x, av0.y), fmaxf(av0.z, av0.w)),
                         fmaxf(fmaxf(av1.x, av1.y), fmaxf(av1.z, av1.w)));
        for (int s = 1; s < 16; s <<= 1)
            pm = fmaxf(pm, __shfl_xor(pm, s, 16));
        if (c16 == 0) s_mrow[row16] = pm;

        // P = exp(alpha - m) in f16 (uses locally reduced pm, no barrier needed)
        v8h ph;
        ph[0] = (_Float16)fast_exp(av0.x - pm);
        ph[1] = (_Float16)fast_exp(av0.y - pm);
        ph[2] = (_Float16)fast_exp(av0.z - pm);
        ph[3] = (_Float16)fast_exp(av0.w - pm);
        ph[4] = (_Float16)fast_exp(av1.x - pm);
        ph[5] = (_Float16)fast_exp(av1.y - pm);
        ph[6] = (_Float16)fast_exp(av1.z - pm);
        ph[7] = (_Float16)fast_exp(av1.w - pm);
        *(v8h*)&s_P[row16][cbase] = ph;
        __syncthreads();

        // old alpha (this thread's own 8 update targets) + per-row maxes
        float oldv[8];
        for (int r = 0; r < 8; ++r) oldv[r] = s_alpha[r + 8 * half][j];
        float4 m0 = *(const float4*)&s_mrow[8 * half];
        float4 m1 = *(const float4*)&s_mrow[8 * half + 4];
        float mr[8] = {m0.x, m0.y, m0.z, m0.w, m1.x, m1.y, m1.z, m1.w};

        // A fragments (all loads issued before the WMMA chain)
        v16h A[4];
        for (int kc = 0; kc < 4; ++kc) {
            int base = kc * 32 + half * 8;
            v8h lo = *(const v8h*)(prow + base);
            v8h hi = *(const v8h*)(prow + base + 16);
            for (int e = 0; e < 8; ++e) { A[kc][e] = lo[e]; A[kc][e + 8] = hi[e]; }
        }
        v8f z = {};
        v8f acc0 = __builtin_amdgcn_wmma_f32_16x16x32_f16(false, A[0], false, Bfrag[0], (short)0, z,    false, false);
        v8f acc1 = __builtin_amdgcn_wmma_f32_16x16x32_f16(false, A[1], false, Bfrag[1], (short)0, z,    false, false);
        acc0     = __builtin_amdgcn_wmma_f32_16x16x32_f16(false, A[2], false, Bfrag[2], (short)0, acc0, false, false);
        acc1     = __builtin_amdgcn_wmma_f32_16x16x32_f16(false, A[3], false, Bfrag[3], (short)0, acc1, false, false);

        // alpha' = feat + m + log(acc), branchless length mask
        for (int r = 0; r < 8; ++r) {
            float v = fv[r] + mr[r] + fast_log(acc0[r] + acc1[r]);
            s_alpha[r + 8 * half][j] = (t < lenr[r]) ? v : oldv[r];
        }
        __syncthreads();
    }

    // ---------------- final: sum_rows logsumexp_j(alpha + trans[j,END]) ----
    float te[8];
    float pm = -3.4e38f;
    for (int e = 0; e < 8; ++e) {
        int c = cbase + e;
        te[e] = s_alpha[row16][c] + trans[c * KTAG + END_TAG];
        pm = fmaxf(pm, te[e]);
    }
    for (int s = 1; s < 16; s <<= 1)
        pm = fmaxf(pm, __shfl_xor(pm, s, 16));
    float ps = 0.0f;
    for (int e = 0; e < 8; ++e) ps += __expf(te[e] - pm);
    for (int s = 1; s < 16; s <<= 1)
        ps += __shfl_xor(ps, s, 16);
    if (c16 == 0) s_res[row16] = __logf(ps) + pm;
    __syncthreads();
    if (tid == 0) {
        float s = 0.0f;
        for (int r = 0; r < 16; ++r) s += s_res[r];
        partial[blockIdx.x] = s;
    }
}

__global__ void crf_reduce_kernel(const float* __restrict__ partial,
                                  float* __restrict__ out, int n)
{
    if (threadIdx.x == 0 && blockIdx.x == 0) {
        float s = 0.0f;
        for (int i = 0; i < n; ++i) s += partial[i];
        out[0] = s;
    }
}

extern "C" void kernel_launch(void* const* d_in, const int* in_sizes, int n_in,
                              void* d_out, int out_size, void* d_ws, size_t ws_size,
                              hipStream_t stream) {
    const float* feats   = (const float*)d_in[0];   // [512,512,128] f32
    const float* trans   = (const float*)d_in[1];   // [128,128] f32
    const int*   lengths = (const int*)d_in[2];     // [512] i32
    float* partial = (float*)d_ws;                  // 32 floats of scratch

    const int nblocks = 512 / 16;                   // 32 row-tiles of 16
    crf_forward_kernel<<<nblocks, 256, 0, stream>>>(feats, trans, lengths, partial);
    crf_reduce_kernel<<<1, 32, 0, stream>>>(partial, (float*)d_out, nblocks);
}
